// VQVAE_56410100466034
// MI455X (gfx1250) — matrix operand order repaired and checked by reading, not compiled, not run on hardware
//
#include <hip/hip_runtime.h>

typedef __attribute__((ext_vector_type(16))) _Float16 v16h;
typedef __attribute__((ext_vector_type(8)))  _Float16 v8h;
typedef __attribute__((ext_vector_type(8)))  float    v8f;

#define N_TOK 32768   // B*T
#define F_DIM 80
#define D_DIM 512
#define K_CENT 2048

// ---------------------------------------------------------------------------
// Kernel A: encoder  z = x @ W_enc + b_enc   (fp32 math, f16 output for WMMA)
// block = 256 threads handles 16 rows x 512 cols
// ---------------------------------------------------------------------------
__global__ __launch_bounds__(256) void encode_kernel(
    const float* __restrict__ x, const float* __restrict__ W_enc,
    const float* __restrict__ b_enc, _Float16* __restrict__ z16) {
  __shared__ float ldsx[16 * F_DIM];
  const int t = threadIdx.x;
  const int n0 = blockIdx.x * 16;

  for (int i = t; i < 16 * F_DIM; i += 256) {
    int r = i / F_DIM, f = i % F_DIM;
    ldsx[i] = x[(size_t)(n0 + r) * F_DIM + f];
  }
  __syncthreads();

  float acc0[16], acc1[16];
#pragma unroll
  for (int r = 0; r < 16; ++r) { acc0[r] = 0.f; acc1[r] = 0.f; }

  const int c0 = t;  // 0..255
  for (int f = 0; f < F_DIM; ++f) {
    float w0 = W_enc[f * D_DIM + c0];
    float w1 = W_enc[f * D_DIM + c0 + 256];
#pragma unroll
    for (int r = 0; r < 16; ++r) {
      float xv = ldsx[r * F_DIM + f];
      acc0[r] += xv * w0;
      acc1[r] += xv * w1;
    }
  }
  float be0 = b_enc[c0], be1 = b_enc[c0 + 256];
#pragma unroll
  for (int r = 0; r < 16; ++r) {
    z16[(size_t)(n0 + r) * D_DIM + c0]       = (_Float16)(acc0[r] + be0);
    z16[(size_t)(n0 + r) * D_DIM + c0 + 256] = (_Float16)(acc1[r] + be1);
  }
}

// ---------------------------------------------------------------------------
// Kernel B: centroid prep — f16 copy + cnorm[k] = ||c_k||^2 (fp32)
// one block per centroid row
// ---------------------------------------------------------------------------
__global__ __launch_bounds__(256) void cprep_kernel(
    const float* __restrict__ centroids, _Float16* __restrict__ c16,
    float* __restrict__ cnorm) {
  __shared__ float red[256];
  const int t = threadIdx.x;
  const int k = blockIdx.x;
  float v0 = centroids[(size_t)k * D_DIM + t];
  float v1 = centroids[(size_t)k * D_DIM + t + 256];
  c16[(size_t)k * D_DIM + t]       = (_Float16)v0;
  c16[(size_t)k * D_DIM + t + 256] = (_Float16)v1;
  red[t] = v0 * v0 + v1 * v1;
  __syncthreads();
  for (int s = 128; s > 0; s >>= 1) {
    if (t < s) red[t] += red[t + s];
    __syncthreads();
  }
  if (t == 0) cnorm[k] = red[0];
}

// ---------------------------------------------------------------------------
// Kernel C: fused distance matmul + argmin via v_wmma_f32_16x16x32_f16.
// argmin_k ||z-c_k||^2  ==  argmax_k ( z.c_k - 0.5*||c_k||^2 )
// One wave owns 32 rows x 64 centroids (2 M-tiles x 4 N-tiles), loops over
// the 2048 centroids in 32 groups, K=512 in 16 steps of 32.
// ---------------------------------------------------------------------------
__global__ __launch_bounds__(256) void vq_argmin_kernel(
    const _Float16* __restrict__ z16, const _Float16* __restrict__ c16,
    const float* __restrict__ cnorm, int* __restrict__ outIdx) {
  const int lane = threadIdx.x & 31;
  const int wave = threadIdx.x >> 5;
  const int w = blockIdx.x * 8 + wave;  // global wave id, 1024 total
  const int r0 = w * 32;                // 32 rows per wave
  const int col = lane & 15;
  const int half = lane >> 4;

  // A fragment layout (16-bit 16x32): lanes 0-15 carry K{0..7,16..23},
  // lanes 16-31 carry K{8..15,24..31} of row (lane&15).
  const int asub = half * 8;
  // B fragment layout (32x16): lane = column, lanes 0-15 K{0..7,16..23},
  // lanes 16-31 K{8..15,24..31} (same order as A).
  const int bsub = half * 8;

  const _Float16* zrow0 = z16 + (size_t)(r0 + col) * D_DIM;
  const _Float16* zrow1 = z16 + (size_t)(r0 + 16 + col) * D_DIM;

  float best[2][8];
  int bidx[2][8];
#pragma unroll
  for (int p = 0; p < 2; ++p)
#pragma unroll
    for (int j = 0; j < 8; ++j) { best[p][j] = -3.4e38f; bidx[p][j] = 0; }

  const v8f vzero = {0.f, 0.f, 0.f, 0.f, 0.f, 0.f, 0.f, 0.f};

  for (int g = 0; g < 32; ++g) {  // 32 groups of 64 centroids
    const int cbase = g * 64;
    v8f acc[2][4];
#pragma unroll
    for (int p = 0; p < 2; ++p)
#pragma unroll
      for (int t = 0; t < 4; ++t) acc[p][t] = vzero;

    for (int kk = 0; kk < 16; ++kk) {  // K dimension: 16 steps of 32
      const int kb = kk * 32;
      v8h a0lo = *(const v8h*)(zrow0 + kb + asub);
      v8h a0hi = *(const v8h*)(zrow0 + kb + 16 + asub);
      v8h a1lo = *(const v8h*)(zrow1 + kb + asub);
      v8h a1hi = *(const v8h*)(zrow1 + kb + 16 + asub);
      v16h a0 = __builtin_shufflevector(a0lo, a0hi, 0, 1, 2, 3, 4, 5, 6, 7, 8,
                                        9, 10, 11, 12, 13, 14, 15);
      v16h a1 = __builtin_shufflevector(a1lo, a1hi, 0, 1, 2, 3, 4, 5, 6, 7, 8,
                                        9, 10, 11, 12, 13, 14, 15);
#pragma unroll
      for (int t = 0; t < 4; ++t) {
        const _Float16* crow = c16 + (size_t)(cbase + t * 16 + col) * D_DIM;
        v8h blo = *(const v8h*)(crow + kb + bsub);
        v8h bhi = *(const v8h*)(crow + kb + bsub + 16);
        v16h b = __builtin_shufflevector(blo, bhi, 0, 1, 2, 3, 4, 5, 6, 7, 8,
                                         9, 10, 11, 12, 13, 14, 15);
        acc[0][t] = __builtin_amdgcn_wmma_f32_16x16x32_f16(
            false, a0, false, b, (short)0, acc[0][t], false, false);
        acc[1][t] = __builtin_amdgcn_wmma_f32_16x16x32_f16(
            false, a1, false, b, (short)0, acc[1][t], false, false);
      }
    }

    // Epilogue: fold in -0.5*||c||^2 and update running argmax.
    // C/D layout: VGPR j, lanes 0-15 -> row j, lanes 16-31 -> row j+8; col = lane&15.
#pragma unroll
    for (int t = 0; t < 4; ++t) {
      const int ci = cbase + t * 16 + col;
      const float sb = -0.5f * cnorm[ci];
#pragma unroll
      for (int p = 0; p < 2; ++p) {
#pragma unroll
        for (int j = 0; j < 8; ++j) {
          float m = acc[p][t][j] + sb;
          if (m > best[p][j]) { best[p][j] = m; bidx[p][j] = ci; }
        }
      }
    }
  }

  // Cross-lane argmax over the 16 columns of each lane-half
  // (xor masks < 16 keep lanes within their half). First-index tie-break.
#pragma unroll
  for (int p = 0; p < 2; ++p) {
#pragma unroll
    for (int j = 0; j < 8; ++j) {
#pragma unroll
      for (int off = 8; off >= 1; off >>= 1) {
        float ob = __shfl_xor(best[p][j], off, 32);
        int oi = __shfl_xor(bidx[p][j], off, 32);
        if (ob > best[p][j] || (ob == best[p][j] && oi < bidx[p][j])) {
          best[p][j] = ob;
          bidx[p][j] = oi;
        }
      }
    }
  }
  if (col == 0) {
#pragma unroll
    for (int p = 0; p < 2; ++p)
#pragma unroll
      for (int j = 0; j < 8; ++j)
        outIdx[r0 + p * 16 + half * 8 + j] = bidx[p][j];
  }
}

// ---------------------------------------------------------------------------
// Kernel D: gather + decoder  out = centroids[idx] @ W_dec + b_dec  (fp32)
// block = 256 threads handles 16 rows x 80 cols (5 outputs/thread)
// ---------------------------------------------------------------------------
__global__ __launch_bounds__(256) void decode_kernel(
    const int* __restrict__ idx, const float* __restrict__ centroids,
    const float* __restrict__ W_dec, const float* __restrict__ b_dec,
    float* __restrict__ out) {
  __shared__ float ldsC[16 * D_DIM];
  const int t = threadIdx.x;
  const int r0 = blockIdx.x * 16;
  const int rl = t >> 4;  // 0..15 local row
  const int cl = t & 15;  // 0..15 column phase

  const int cidx = idx[r0 + rl];
  const float* crow = centroids + (size_t)cidx * D_DIM;
  for (int d = cl; d < D_DIM; d += 16) ldsC[rl * D_DIM + d] = crow[d];
  __syncthreads();

  float acc[5];
#pragma unroll
  for (int i = 0; i < 5; ++i) acc[i] = 0.f;

  const float* cr = &ldsC[rl * D_DIM];
  for (int d = 0; d < D_DIM; ++d) {
    float cv = cr[d];
#pragma unroll
    for (int i = 0; i < 5; ++i) acc[i] += cv * W_dec[d * F_DIM + cl + 16 * i];
  }
  float* orow = out + (size_t)(r0 + rl) * F_DIM;
#pragma unroll
  for (int i = 0; i < 5; ++i) orow[cl + 16 * i] = acc[i] + b_dec[cl + 16 * i];
}

// ---------------------------------------------------------------------------
extern "C" void kernel_launch(void* const* d_in, const int* in_sizes, int n_in,
                              void* d_out, int out_size, void* d_ws,
                              size_t ws_size, hipStream_t stream) {
  const float* x         = (const float*)d_in[0];
  const float* W_enc     = (const float*)d_in[1];
  const float* b_enc     = (const float*)d_in[2];
  const float* centroids = (const float*)d_in[3];
  const float* W_dec     = (const float*)d_in[4];
  const float* b_dec     = (const float*)d_in[5];
  float* out = (float*)d_out;

  char* ws = (char*)d_ws;
  _Float16* z16 = (_Float16*)ws;                         // 32 MB
  _Float16* c16 = (_Float16*)(ws + (size_t)33554432);    // 2 MB
  float* cnorm  = (float*)(ws + (size_t)35651584);       // 8 KB
  int* aidx     = (int*)(ws + (size_t)35659776);         // 128 KB

  encode_kernel<<<N_TOK / 16, 256, 0, stream>>>(x, W_enc, b_enc, z16);
  cprep_kernel<<<K_CENT, 256, 0, stream>>>(centroids, c16, cnorm);
  vq_argmin_kernel<<<N_TOK / 32 / 8, 256, 0, stream>>>(z16, c16, cnorm, aidx);
  decode_kernel<<<N_TOK / 16, 256, 0, stream>>>(aidx, centroids, W_dec, b_dec,
                                                out);
}
